// MixResidualBlockC_64115271794981
// MI455X (gfx1250) — matrix operand order repaired and checked by reading, not compiled, not run on hardware
//
#include <hip/hip_runtime.h>
#include <hip/hip_bf16.h>

typedef _Float16 f16;
typedef __attribute__((ext_vector_type(16))) _Float16 v16h;
typedef __attribute__((ext_vector_type(8)))  _Float16 v8h;
typedef __attribute__((ext_vector_type(8)))  float    v8f;
typedef __attribute__((ext_vector_type(4)))  int      v4i;

#define NB   16
#define FIN  256
#define FOUT 256
#define FH2  256
#define NMIX 8
#define LATS 512
#define HWPX 4096

#define AS_GLOBAL __attribute__((address_space(1)))
#define AS_LDS    __attribute__((address_space(3)))

#if defined(__has_builtin)
#if __has_builtin(__builtin_amdgcn_global_load_async_to_lds_b128)
#define USE_ASYNC_LDS 1
#endif
#endif

static __device__ inline v16h cat16(v8h lo, v8h hi) {
  return __builtin_shufflevector(lo, hi, 0,1,2,3,4,5,6,7,8,9,10,11,12,13,14,15);
}

static __device__ inline void async_copy16(const f16* g, f16* l) {
#ifdef USE_ASYNC_LDS
  __builtin_amdgcn_global_load_async_to_lds_b128((AS_GLOBAL v4i*)g, (AS_LDS v4i*)l, 0, 0);
#else
  *(uint4*)l = *(const uint4*)g;
#endif
}

static __device__ inline void async_fence() {
#ifdef USE_ASYNC_LDS
#if __has_builtin(__builtin_amdgcn_s_wait_asynccnt)
  __builtin_amdgcn_s_wait_asynccnt(0);
#else
  asm volatile("s_wait_asynccnt 0x0" ::: "memory");
#endif
#endif
}

// ---------------- mixture coefficients: mix[b][m] = b_dyna[m] + lat[b]·w_dyna[m]
__global__ void mix_coeff_kernel(float* __restrict__ mix, const float* __restrict__ lat,
                                 const float* __restrict__ wd, const float* __restrict__ bd) {
  int t = threadIdx.x;
  if (t >= NB * NMIX) return;
  int b = t / NMIX, m = t % NMIX;
  float a = bd[m];
  const float* lr = lat + b * LATS;
  const float* wr = wd + m * LATS;
  for (int l = 0; l < LATS; ++l) a = fmaf(lr[l], wr[l], a);
  mix[t] = a;
}

// ---------------- per-sample weights (f16) / biases (f32): dst[b][e] = sum_m mix[b][m]*src[m][e]
__global__ void mix_apply_f16_kernel(f16* __restrict__ dst, const float* __restrict__ src,
                                     const float* __restrict__ mix, long E) {
  long g = (long)blockIdx.x * blockDim.x + threadIdx.x;
  if (g >= (long)NB * E) return;
  int b = (int)(g / E);
  long e = g - (long)b * E;
  const float* m = mix + b * NMIX;
  float a = 0.f;
#pragma unroll
  for (int j = 0; j < NMIX; ++j) a = fmaf(m[j], src[(long)j * E + e], a);
  dst[g] = (f16)a;
}

__global__ void mix_apply_f32_kernel(float* __restrict__ dst, const float* __restrict__ src,
                                     const float* __restrict__ mix, long E) {
  long g = (long)blockIdx.x * blockDim.x + threadIdx.x;
  if (g >= (long)NB * E) return;
  int b = (int)(g / E);
  long e = g - (long)b * E;
  const float* m = mix + b * NMIX;
  float a = 0.f;
#pragma unroll
  for (int j = 0; j < NMIX; ++j) a = fmaf(m[j], src[(long)j * E + e], a);
  dst[g] = a;
}

// ---------------- x [B][FIN][HW] f32 -> xT [B][HW][FIN] f16 (LDS tiled transpose)
__global__ void transpose_x_kernel(f16* __restrict__ xT, const float* __restrict__ x) {
  __shared__ float tile[32][33];
  int p0 = blockIdx.x * 32, c0 = blockIdx.y * 32, b = blockIdx.z;
  int tx = threadIdx.x, ty = threadIdx.y;
#pragma unroll
  for (int i = 0; i < 4; ++i) {
    int c = c0 + ty + i * 8;
    tile[ty + i * 8][tx] = x[((size_t)b * FIN + c) * HWPX + p0 + tx];
  }
  __syncthreads();
#pragma unroll
  for (int i = 0; i < 4; ++i) {
    int p = p0 + ty + i * 8;
    xT[((size_t)b * HWPX + p) * FIN + c0 + tx] = (f16)tile[tx][ty + i * 8];
  }
}

// ---------------- fused per-sample GEMM stage
// C[64 px][256 outch] = A1[px][K1] * W1[outch][K1]^T  (+ A2*W2^T if K2>0) + bias
// MODE 0: write sin/cos f16 to out[b][px][2*FH2] (concat: sin at c, cos at FH2+c)
// MODE 1: write f32 to out[b][outch][HW] (channel-major, via LDS transpose)
// WMMA orientation: A (M=pixel, K=channel) from LDS, B (N=outch, K) from global.
template <int K1, int K2, int MODE>
__global__ __launch_bounds__(256) void stage_gemm_kernel(
    const f16* __restrict__ W1, const f16* __restrict__ A1g, const float* __restrict__ bias1,
    const f16* __restrict__ W2, const f16* __restrict__ A2g, const float* __restrict__ bias2,
    void* __restrict__ outp) {
  constexpr int K1P = K1 + 8;
  constexpr int K2P = (K2 > 0) ? (K2 + 8) : 0;
  __shared__ __align__(16) f16 lds[64 * K1P + 64 * K2P];

  const int tid = threadIdx.x;
  const int b = blockIdx.x >> 6;
  const int P0 = (blockIdx.x & 63) * 64;

  // stage activation tiles into LDS (16B chunks, K-major rows, +8-half pad)
  {
    const f16* g = A1g + ((size_t)b * HWPX + P0) * K1;
    constexpr int CW = K1 / 8;
    for (int idx = tid; idx < 64 * CW; idx += 256) {
      int p = idx / CW, kc = idx - p * CW;
      async_copy16(g + (size_t)p * K1 + kc * 8, &lds[p * K1P + kc * 8]);
    }
  }
  f16* lds2 = lds + 64 * K1P;
  if constexpr (K2 > 0) {
    const f16* g = A2g + ((size_t)b * HWPX + P0) * K2;
    constexpr int CW = K2 / 8;
    for (int idx = tid; idx < 64 * CW; idx += 256) {
      int p = idx / CW, kc = idx - p * CW;
      async_copy16(g + (size_t)p * K2 + kc * 8, &lds2[p * K2P + kc * 8]);
    }
  }
  async_fence();
  __syncthreads();

  const int w = tid >> 5, lane = tid & 31;
  const int l16 = lane & 15, hs = lane >> 4;
  const int mrow = 32 * (w >> 2);  // pixel offset within tile: 2 M-tiles (+0,+16)
  const int ncol = 64 * (w & 3);   // out-channel offset: 4 N-tiles

  v8f acc[2][4] = {};

  const f16* W1b = W1 + (size_t)b * 256 * K1;
  for (int k0 = 0; k0 < K1; k0 += 32) {
    v16h af[2], bf[4];
#pragma unroll
    for (int mi = 0; mi < 2; ++mi) {
      const f16* p = &lds[(mrow + 16 * mi + l16) * K1P + k0 + 8 * hs];
      af[mi] = cat16(*(const v8h*)p, *(const v8h*)(p + 16));
    }
#pragma unroll
    for (int ni = 0; ni < 4; ++ni) {
      const f16* p = W1b + (size_t)(ncol + 16 * ni + l16) * K1 + k0 + 8 * hs;
      bf[ni] = cat16(*(const v8h*)p, *(const v8h*)(p + 16));
    }
#pragma unroll
    for (int mi = 0; mi < 2; ++mi)
#pragma unroll
      for (int ni = 0; ni < 4; ++ni)
        acc[mi][ni] = __builtin_amdgcn_wmma_f32_16x16x32_f16(
            false, af[mi], false, bf[ni], (short)0, acc[mi][ni], false, false);
  }

  if constexpr (K2 > 0) {
    const f16* W2b = W2 + (size_t)b * 256 * K2;
    for (int k0 = 0; k0 < K2; k0 += 32) {
      v16h af[2], bf[4];
#pragma unroll
      for (int mi = 0; mi < 2; ++mi) {
        const f16* p = &lds2[(mrow + 16 * mi + l16) * K2P + k0 + 8 * hs];
        af[mi] = cat16(*(const v8h*)p, *(const v8h*)(p + 16));
      }
#pragma unroll
      for (int ni = 0; ni < 4; ++ni) {
        const f16* p = W2b + (size_t)(ncol + 16 * ni + l16) * K2 + k0 + 8 * hs;
        bf[ni] = cat16(*(const v8h*)p, *(const v8h*)(p + 16));
      }
#pragma unroll
      for (int mi = 0; mi < 2; ++mi)
#pragma unroll
        for (int ni = 0; ni < 4; ++ni)
          acc[mi][ni] = __builtin_amdgcn_wmma_f32_16x16x32_f16(
              false, af[mi], false, bf[ni], (short)0, acc[mi][ni], false, false);
    }
  }

  float bi[4];
#pragma unroll
  for (int ni = 0; ni < 4; ++ni) {
    int c = ncol + 16 * ni + l16;
    float t = bias1[b * 256 + c];
    if constexpr (K2 > 0) t += bias2[b * 256 + c];
    bi[ni] = t;
  }

  if constexpr (MODE == 0) {
    f16* outb = (f16*)outp + ((size_t)b * HWPX + P0) * (2 * FH2);
#pragma unroll
    for (int mi = 0; mi < 2; ++mi)
#pragma unroll
      for (int ni = 0; ni < 4; ++ni) {
        int c = ncol + 16 * ni + l16;
#pragma unroll
        for (int r = 0; r < 8; ++r) {
          int p = mrow + 16 * mi + 8 * hs + r;
          float v = acc[mi][ni][r] + bi[ni];
          float sv, cv;
          __sincosf(v, &sv, &cv);
          outb[(size_t)p * (2 * FH2) + c] = (f16)sv;
          outb[(size_t)p * (2 * FH2) + FH2 + c] = (f16)cv;
        }
      }
  } else {
    __syncthreads();                 // LDS activation reads done; reuse as f32 out stage
    float* lout = (float*)lds;       // [64][258] floats (66048B <= 64*K1P*2 for K1=512)
#pragma unroll
    for (int mi = 0; mi < 2; ++mi)
#pragma unroll
      for (int ni = 0; ni < 4; ++ni) {
        int c = ncol + 16 * ni + l16;
#pragma unroll
        for (int r = 0; r < 8; ++r) {
          int p = mrow + 16 * mi + 8 * hs + r;
          lout[p * 258 + c] = acc[mi][ni][r] + bi[ni];
        }
      }
    __syncthreads();
    float* outb = (float*)outp + (size_t)b * FOUT * HWPX + P0;
    for (int idx = tid; idx < 64 * FOUT; idx += 256) {
      int c = idx >> 6, p = idx & 63;
      outb[(size_t)c * HWPX + p] = lout[p * 258 + c];
    }
  }
}

extern "C" void kernel_launch(void* const* d_in, const int* in_sizes, int n_in,
                              void* d_out, int out_size, void* d_ws, size_t ws_size,
                              hipStream_t stream) {
  const float* x            = (const float*)d_in[0];
  const float* lat          = (const float*)d_in[1];
  const float* k_in_mix     = (const float*)d_in[2];
  const float* k_mid_mix    = (const float*)d_in[3];
  const float* k_out_mix    = (const float*)d_in[4];
  const float* k_short_mix  = (const float*)d_in[5];
  const float* b_in_mix     = (const float*)d_in[6];
  const float* b_mid_mix    = (const float*)d_in[7];
  const float* b_out_mix    = (const float*)d_in[8];
  const float* b_short_mix  = (const float*)d_in[9];
  const float* w_dyna       = (const float*)d_in[10];
  const float* b_dyna       = (const float*)d_in[11];
  (void)in_sizes; (void)n_in; (void)out_size; (void)ws_size;

  // workspace layout (~108 MB), 256B aligned slots
  char* ws = (char*)d_ws;
  size_t o = 0;
  auto take = [&](size_t bytes) { size_t r = o; o = (o + bytes + 255) & ~(size_t)255; return r; };
  float* mix   = (float*)(ws + take((size_t)NB * NMIX * 4));
  float* b_in  = (float*)(ws + take((size_t)NB * FH2 * 4));
  float* b_mid = (float*)(ws + take((size_t)NB * FH2 * 4));
  float* b_out = (float*)(ws + take((size_t)NB * FOUT * 4));
  float* b_sh  = (float*)(ws + take((size_t)NB * FOUT * 4));
  f16* W_in  = (f16*)(ws + take((size_t)NB * FH2 * FIN * 2));        // 2 MB
  f16* W_mid = (f16*)(ws + take((size_t)NB * FH2 * 2 * FH2 * 2));    // 4 MB
  f16* W_out = (f16*)(ws + take((size_t)NB * FOUT * 2 * FH2 * 2));   // 4 MB
  f16* W_sh  = (f16*)(ws + take((size_t)NB * FOUT * FIN * 2));       // 2 MB
  f16* xT    = (f16*)(ws + take((size_t)NB * HWPX * FIN * 2));       // 32 MB
  f16* hbuf  = (f16*)(ws + take((size_t)NB * HWPX * 2 * FH2 * 2));   // 64 MB (h1, overwritten in place by h2)

  mix_coeff_kernel<<<1, 128, 0, stream>>>(mix, lat, w_dyna, b_dyna);

  auto gs = [](long n) { return (unsigned)((n + 255) / 256); };
  mix_apply_f16_kernel<<<gs((long)NB * FH2 * FIN), 256, 0, stream>>>(W_in, k_in_mix, mix, (long)FH2 * FIN);
  mix_apply_f16_kernel<<<gs((long)NB * FH2 * 2 * FH2), 256, 0, stream>>>(W_mid, k_mid_mix, mix, (long)FH2 * 2 * FH2);
  mix_apply_f16_kernel<<<gs((long)NB * FOUT * 2 * FH2), 256, 0, stream>>>(W_out, k_out_mix, mix, (long)FOUT * 2 * FH2);
  mix_apply_f16_kernel<<<gs((long)NB * FOUT * FIN), 256, 0, stream>>>(W_sh, k_short_mix, mix, (long)FOUT * FIN);
  mix_apply_f32_kernel<<<gs((long)NB * FH2), 256, 0, stream>>>(b_in, b_in_mix, mix, FH2);
  mix_apply_f32_kernel<<<gs((long)NB * FH2), 256, 0, stream>>>(b_mid, b_mid_mix, mix, FH2);
  mix_apply_f32_kernel<<<gs((long)NB * FOUT), 256, 0, stream>>>(b_out, b_out_mix, mix, FOUT);
  mix_apply_f32_kernel<<<gs((long)NB * FOUT), 256, 0, stream>>>(b_sh, b_short_mix, mix, FOUT);

  transpose_x_kernel<<<dim3(HWPX / 32, FIN / 32, NB), dim3(32, 8), 0, stream>>>(xT, x);

  const unsigned NWG = NB * (HWPX / 64);  // 1024 workgroups
  // stage 1: h1 = sincos(W_in · x + b_in)      (K=256)
  stage_gemm_kernel<FIN, 0, 0><<<NWG, 256, 0, stream>>>(W_in, xT, b_in, nullptr, nullptr, nullptr, hbuf);
  // stage 2: h2 = sincos(W_mid · h1 + b_mid)   (K=512, in place)
  stage_gemm_kernel<2 * FH2, 0, 0><<<NWG, 256, 0, stream>>>(W_mid, hbuf, b_mid, nullptr, nullptr, nullptr, hbuf);
  // stage 3: out = W_out·h2 + W_short·x + b_out + b_short   (K=512 + K=256)
  stage_gemm_kernel<2 * FH2, FIN, 1><<<NWG, 256, 0, stream>>>(W_out, hbuf, b_out, W_sh, xT, b_sh, d_out);
}
